// Allegro_78134045049177
// MI455X (gfx1250) — matrix operand order repaired and checked by reading, not compiled, not run on hardware
//
#include <hip/hip_runtime.h>

// ---------------------------------------------------------------------------
// Allegro edge-network forward for MI455X (gfx1250, wave32, WMMA).
// 1 wave = 16 edges. All dense layers run on v_wmma_f32_16x16x32_f16 with
// f32 accumulation; weights pre-staged to LDS in WMMA B-fragment order,
// activations staged to LDS in WMMA A-fragment order. Small SO(3) tensor
// contractions run in f32 VALU with registers only.
// ---------------------------------------------------------------------------

typedef _Float16 h16 __attribute__((ext_vector_type(16)));
typedef float    f8  __attribute__((ext_vector_type(8)));
typedef unsigned int u4 __attribute__((ext_vector_type(4)));

// DS ops from one wave execute in-order; this waits them out and acts as a
// compiler memory barrier for our wave-private LDS staging regions.
#define WSYNC() asm volatile("s_wait_dscnt 0" ::: "memory")

// ---- LDS layout (halves for weights) --------------------------------------
static constexpr int OFF_WE1   = 0;            // 64n x 32k
static constexpr int OFF_WE2   = 2048;         // 64n x 64k
static constexpr int OFF_WENVE = 6144;         // 16n x 64k
static constexpr int OFF_L0    = 7168;
static constexpr int LSTRIDE   = 14336;        // per-layer block
static constexpr int L_LIN0 = 0;               // 16n x 64k
static constexpr int L_LIN1 = 1024;            // 16n x 64k
static constexpr int L_LIN2 = 2048;            // 16n x 64k
static constexpr int L_M1   = 3072;            // 64n x 96k
static constexpr int L_M2   = 9216;            // 64n x 64k
static constexpr int L_ENV  = 13312;           // 16n x 64k
static constexpr int WGT_HALVES = OFF_L0 + 2 * LSTRIDE;   // 35840
// per-wave byte offsets inside the wave scratch block
static constexpr int PW_S    = 0;              // scal f16   16x64  (2048 B)
static constexpr int PW_A    = 2048;           // A stage    16x96  (3072 B)
static constexpr int PW_N    = 5120;           // n tiles f16 9x16x16 (4608 B)
static constexpr int PW_GEO  = 9728;           // y1,y2 f32  16x8   (512 B)
static constexpr int PW_ENV  = 10240;          // env f32    16x16  (1024 B)
static constexpr int PER_WAVE_B = 11264;
static constexpr int SMEM_B = WGT_HALVES * 2 + 4 * PER_WAVE_B;   // 116736 B

// ---- small helpers --------------------------------------------------------
__device__ __forceinline__ f8 zero8() {
    f8 z;
#pragma unroll
    for (int i = 0; i < 8; ++i) z[i] = 0.0f;
    return z;
}

// A fragment: activations staged row-major [16 rows][stride halves].
// 16-bit A 16x32 layout: lane row = lane&15, halves e<8 -> K=k0+8g+e,
// e>=8 -> K=k0+16+8g+(e-8)  => two contiguous 16-byte runs.
__device__ __forceinline__ h16 afrag(const _Float16* base, int stride, int k0, int lane) {
    const int row = lane & 15, g = lane >> 4;
    union { h16 v; u4 u[2]; } f;
    f.u[0] = *(const u4*)(base + row * stride + k0 + 8 * g);
    f.u[1] = *(const u4*)(base + row * stride + k0 + 16 + 8 * g);
    return f.v;
}

// B fragment: weight tile staged column-major [16 cols][kpad halves].
// 16-bit B layout: lane col = lane&15, halves e<8 -> K=k0+8g+e,
// e>=8 -> K=k0+16+8g+(e-8)  => two contiguous 16-byte runs.
__device__ __forceinline__ h16 bfrag(const _Float16* tile, int kpad, int k0, int lane) {
    const int n = lane & 15, g = lane >> 4;
    union { h16 v; u4 u[2]; } f;
    f.u[0] = *(const u4*)(tile + n * kpad + k0 + 8 * g);
    f.u[1] = *(const u4*)(tile + n * kpad + k0 + 16 + 8 * g);
    return f.v;
}

__device__ __forceinline__ f8 wmma(h16 a, h16 b, f8 c) {
    return __builtin_amdgcn_wmma_f32_16x16x32_f16(false, a, false, b, (short)0, c, false, false);
}

__device__ __forceinline__ float silu_f(float x) {
    return x / (1.0f + __expf(-x));
}

// Q-tensor helpers.  Q[m] are symmetric 3x3 generators.
__device__ __forceinline__ void mat5(const float a[5], float M[9]) {
    const float S2 = 0.70710678118654752f, R6 = 0.40824829046386302f;
    M[0] = -a[2] * R6 + a[4] * S2;
    M[4] = -a[2] * R6 - a[4] * S2;
    M[8] = 2.0f * a[2] * R6;
    M[1] = M[3] = a[0] * S2;
    M[5] = M[7] = a[1] * S2;
    M[2] = M[6] = a[3] * S2;
}
__device__ __forceinline__ void proj5(const float M[9], float p[5]) {
    const float S2 = 0.70710678118654752f, R6 = 0.40824829046386302f;
    p[0] = S2 * (M[1] + M[3]);
    p[1] = S2 * (M[5] + M[7]);
    p[2] = R6 * (2.0f * M[8] - M[0] - M[4]);
    p[3] = S2 * (M[2] + M[6]);
    p[4] = S2 * (M[0] - M[4]);
}
__device__ __forceinline__ void mm3(const float A[9], const float B[9], float C[9]) {
#pragma unroll
    for (int i = 0; i < 3; ++i)
#pragma unroll
        for (int j = 0; j < 3; ++j)
            C[i * 3 + j] = A[i * 3 + 0] * B[0 * 3 + j] + A[i * 3 + 1] * B[1 * 3 + j] + A[i * 3 + 2] * B[2 * 3 + j];
}

// Stage a (K x N) row-major f32 weight into LDS f16, column-major with K
// padded to kpad (zeros), tiles of 16 columns contiguous.
__device__ __forceinline__ void stage_w(const float* W, _Float16* dst,
                                        int K, int N, int kpad, int tid, int nth) {
    for (int idx = tid; idx < N * kpad; idx += nth) {
        int n = idx / kpad, k = idx - n * kpad;
        dst[n * kpad + k] = (_Float16)((k < K) ? W[k * N + n] : 0.0f);
    }
}

__global__ __launch_bounds__(128)
void allegro_fwd(const float* __restrict__ bond_dist, const float* __restrict__ bond_diff,
                 const float* __restrict__ emb, const float* __restrict__ W_e1,
                 const float* __restrict__ b_e1, const float* __restrict__ W_e2,
                 const float* __restrict__ b_e2, const float* __restrict__ W_env_e,
                 const float* __restrict__ b_env_e, const float* __restrict__ w_tp,
                 const float* __restrict__ Wlin0, const float* __restrict__ Wlin1,
                 const float* __restrict__ Wlin2, const float* __restrict__ Wm1,
                 const float* __restrict__ bm1, const float* __restrict__ Wm2,
                 const float* __restrict__ bm2, const float* __restrict__ Wenv,
                 const float* __restrict__ benv, const int* __restrict__ Z,
                 const int* __restrict__ ei, float* __restrict__ out)
{
    extern __shared__ char smem[];
    _Float16* wlds = (_Float16*)smem;

    const int tid = threadIdx.x, nth = blockDim.x;
    const int wave = threadIdx.x >> 5;
    const int lane = threadIdx.x & 31;

    if (tid == 0) {
        __builtin_prefetch(W_e2, 0, 1);
        __builtin_prefetch(Wm1, 0, 1);
    }

    // ---- stage all weights to LDS (f16, B-fragment order) -----------------
    stage_w(W_e1,   wlds + OFF_WE1,   24, 64, 32, tid, nth);
    stage_w(W_e2,   wlds + OFF_WE2,   64, 64, 64, tid, nth);
    stage_w(W_env_e, wlds + OFF_WENVE, 64, 16, 64, tid, nth);
#pragma unroll
    for (int l = 0; l < 2; ++l) {
        _Float16* base = wlds + OFF_L0 + l * LSTRIDE;
        stage_w(Wlin0 + l * 48 * 16, base + L_LIN0, 48, 16, 64, tid, nth);
        stage_w(Wlin1 + l * 64 * 16, base + L_LIN1, 64, 16, 64, tid, nth);
        stage_w(Wlin2 + l * 64 * 16, base + L_LIN2, 64, 16, 64, tid, nth);
        stage_w(Wm1   + l * 80 * 64, base + L_M1,   80, 64, 96, tid, nth);
        stage_w(Wm2   + l * 64 * 64, base + L_M2,   64, 64, 64, tid, nth);
        stage_w(Wenv  + l * 64 * 16, base + L_ENV,  64, 16, 64, tid, nth);
    }
    __syncthreads();

    // per-wave scratch
    char* my = smem + WGT_HALVES * 2 + wave * PER_WAVE_B;
    _Float16* sS   = (_Float16*)(my + PW_S);    // scal f16 [16][64]
    _Float16* aS   = (_Float16*)(my + PW_A);    // A stage  [16][96]
    _Float16* nS   = (_Float16*)(my + PW_N);    // n tiles  [9][16][16]
    float*    geo  = (float*)   (my + PW_GEO);  // [16][8]  y1(3), y2(5)
    float*    envS = (float*)   (my + PW_ENV);  // [16][16]

    const int e0 = blockIdx.x * 64 + wave * 16;
    const int el = lane & 15, g = lane >> 4, cb = g * 8;

    // ---- phase 0: per-edge radial basis, geometry, pair embedding ---------
    if (lane < 16) {
        const int e = e0 + lane;
        const float bd = bond_dist[e];
        const float dx = bond_diff[3 * e + 0];
        const float dy = bond_diff[3 * e + 1];
        const float dz = bond_diff[3 * e + 2];
        const float u = bd * 0.2f;
        const float u2 = u * u, u3 = u2 * u, u6 = u3 * u3, u7 = u6 * u, u8 = u7 * u;
        const float cut = (u < 1.0f) ? (1.0f - 28.0f * u6 + 48.0f * u7 - 21.0f * u8) : 0.0f;
        const float pref = 0.63245553203f /*sqrt(2/5)*/ / bd * cut;
#pragma unroll
        for (int nb = 0; nb < 8; ++nb)
            aS[lane * 96 + nb] = (_Float16)(pref * __sinf(3.14159265358979f * (float)(nb + 1) * u));
        const float inv = 1.0f / (bd + 1e-8f);
        const float d0 = dx * inv, d1 = dy * inv, d2 = dz * inv;
        geo[lane * 8 + 0] = d0;
        geo[lane * 8 + 1] = d1;
        geo[lane * 8 + 2] = d2;
        float O[9] = { d0 * d0, d0 * d1, d0 * d2,
                       d1 * d0, d1 * d1, d1 * d2,
                       d2 * d0, d2 * d1, d2 * d2 };
        float p[5];
        proj5(O, p);
#pragma unroll
        for (int m = 0; m < 5; ++m) geo[lane * 8 + 3 + m] = 1.22474487139f /*sqrt(1.5)*/ * p[m];
        const int zs = Z[ei[2 * e + 0]], zd = Z[ei[2 * e + 1]];
#pragma unroll
        for (int c = 0; c < 16; ++c)
            aS[lane * 96 + 8 + c] = (_Float16)(emb[zs * 16 + c] * emb[zd * 16 + c]);
#pragma unroll
        for (int k = 24; k < 32; ++k) aS[lane * 96 + k] = (_Float16)0.0f;
    }
    WSYNC();

    f8 acc[4];

    // ---- scal = silu(silu(h @ W_e1 + b1) @ W_e2 + b2) ---------------------
    {
        h16 a = afrag(aS, 96, 0, lane);
#pragma unroll
        for (int nt = 0; nt < 4; ++nt)
            acc[nt] = wmma(a, bfrag(wlds + OFF_WE1 + nt * 16 * 32, 32, 0, lane), zero8());
    }
#pragma unroll
    for (int nt = 0; nt < 4; ++nt) {
        const float bb = b_e1[nt * 16 + el];
#pragma unroll
        for (int r = 0; r < 8; ++r)
            aS[(r + 8 * g) * 96 + nt * 16 + el] = (_Float16)silu_f(acc[nt][r] + bb);
    }
    WSYNC();
#pragma unroll
    for (int nt = 0; nt < 4; ++nt) acc[nt] = zero8();
#pragma unroll
    for (int kc = 0; kc < 2; ++kc) {
        h16 a = afrag(aS, 96, kc * 32, lane);
#pragma unroll
        for (int nt = 0; nt < 4; ++nt)
            acc[nt] = wmma(a, bfrag(wlds + OFF_WE2 + nt * 16 * 64, 64, kc * 32, lane), acc[nt]);
    }
#pragma unroll
    for (int nt = 0; nt < 4; ++nt) {
        const float bb = b_e2[nt * 16 + el];
#pragma unroll
        for (int r = 0; r < 8; ++r)
            sS[(r + 8 * g) * 64 + nt * 16 + el] = (_Float16)silu_f(acc[nt][r] + bb);
    }
    WSYNC();

    // ---- env = scal @ W_env_e + b_env_e -----------------------------------
    {
        f8 ea = zero8();
#pragma unroll
        for (int kc = 0; kc < 2; ++kc)
            ea = wmma(afrag(sS, 64, kc * 32, lane),
                      bfrag(wlds + OFF_WENVE, 64, kc * 32, lane), ea);
        const float bb = b_env_e[el];
#pragma unroll
        for (int r = 0; r < 8; ++r) envS[(r + 8 * g) * 16 + el] = ea[r] + bb;
    }
    WSYNC();

    // ---- per-lane register state: edge el, channels cb..cb+7 --------------
    float y1v[3], y2v[5];
#pragma unroll
    for (int i = 0; i < 3; ++i) y1v[i] = geo[el * 8 + i];
#pragma unroll
    for (int m = 0; m < 5; ++m) y2v[m] = geo[el * 8 + 3 + m];

    float envr[8], x0r[8], x1r[3][8], x2r[5][8];
#pragma unroll
    for (int cc = 0; cc < 8; ++cc) {
        envr[cc] = envS[el * 16 + cb + cc];
        x0r[cc] = envr[cc];
#pragma unroll
        for (int i = 0; i < 3; ++i) x1r[i][cc] = y1v[i] * envr[cc];
#pragma unroll
        for (int m = 0; m < 5; ++m) x2r[m][cc] = y2v[m] * envr[cc];
    }

    float My2[9];
    mat5(y2v, My2);

    const float i48 = 0.14433756729740644f;   // 1/sqrt(48)
    const float i64 = 0.125f;                 // 1/sqrt(64)

    for (int l = 0; l < 2; ++l) {
        const _Float16* wl = wlds + OFF_L0 + l * LSTRIDE;
        const float* wt = w_tp + l * 11 * 16;

        // ---- z0 tile: [t000 | t110 | t220 | 0] ----------------------------
#pragma unroll
        for (int cc = 0; cc < 8; ++cc) {
            const int c = cb + cc;
            const float t000 = x0r[cc] * wt[0 * 16 + c];
            const float t110 = (x1r[0][cc] * y1v[0] + x1r[1][cc] * y1v[1] + x1r[2][cc] * y1v[2]) * wt[4 * 16 + c];
            float s220 = 0.0f;
#pragma unroll
            for (int m = 0; m < 5; ++m) s220 += x2r[m][cc] * y2v[m];
            const float t220 = s220 * wt[9 * 16 + c];
            aS[el * 96 + 0  + c] = (_Float16)t000;
            aS[el * 96 + 16 + c] = (_Float16)t110;
            aS[el * 96 + 32 + c] = (_Float16)t220;
            aS[el * 96 + 48 + c] = (_Float16)0.0f;   // z0 K-pad
            aS[el * 96 + 80 + c] = (_Float16)0.0f;   // comb K-pad strip
        }
        WSYNC();
        {
            f8 na = zero8();
#pragma unroll
            for (int kc = 0; kc < 2; ++kc)
                na = wmma(afrag(aS, 96, kc * 32, lane),
                          bfrag(wl + L_LIN0, 64, kc * 32, lane), na);
#pragma unroll
            for (int r = 0; r < 8; ++r) {
                const _Float16 hv = (_Float16)(na[r] * i48);
                nS[0 * 256 + (r + 8 * g) * 16 + el] = hv;      // n0 (next-layer x0)
                aS[(r + 8 * g) * 96 + 64 + el] = hv;           // inv part of comb
            }
        }
        WSYNC();

        // ---- z1 tiles i=0..2: [t011 | t101 | t121 | t211] -----------------
#pragma unroll
        for (int i = 0; i < 3; ++i) {
#pragma unroll
            for (int cc = 0; cc < 8; ++cc) {
                const int c = cb + cc;
                float x2c[5];
#pragma unroll
                for (int m = 0; m < 5; ++m) x2c[m] = x2r[m][cc];
                const float t011 = x0r[cc] * y1v[i] * wt[1 * 16 + c];
                const float t101 = x1r[i][cc] * wt[3 * 16 + c];
                const float t121 = (My2[3 * i + 0] * x1r[0][cc] + My2[3 * i + 1] * x1r[1][cc] +
                                    My2[3 * i + 2] * x1r[2][cc]) * wt[6 * 16 + c];
                float Mx2[9];
                mat5(x2c, Mx2);
                const float t211 = (Mx2[3 * i + 0] * y1v[0] + Mx2[3 * i + 1] * y1v[1] +
                                    Mx2[3 * i + 2] * y1v[2]) * wt[8 * 16 + c];
                aS[el * 96 + 0  + c] = (_Float16)t011;
                aS[el * 96 + 16 + c] = (_Float16)t101;
                aS[el * 96 + 32 + c] = (_Float16)t121;
                aS[el * 96 + 48 + c] = (_Float16)t211;
            }
            WSYNC();
            f8 na = zero8();
#pragma unroll
            for (int kc = 0; kc < 2; ++kc)
                na = wmma(afrag(aS, 96, kc * 32, lane),
                          bfrag(wl + L_LIN1, 64, kc * 32, lane), na);
#pragma unroll
            for (int r = 0; r < 8; ++r)
                nS[(1 + i) * 256 + (r + 8 * g) * 16 + el] = (_Float16)(na[r] * i64);
            WSYNC();
        }

        // ---- z2 tiles m=0..4: [t022 | t112 | t202 | t222] -----------------
        float t112v[5][8], t222v[5][8];
#pragma unroll
        for (int cc = 0; cc < 8; ++cc) {
            float x1c[3] = { x1r[0][cc], x1r[1][cc], x1r[2][cc] };
            float x2c[5];
#pragma unroll
            for (int m = 0; m < 5; ++m) x2c[m] = x2r[m][cc];
            float O[9] = { x1c[0] * y1v[0], x1c[0] * y1v[1], x1c[0] * y1v[2],
                           x1c[1] * y1v[0], x1c[1] * y1v[1], x1c[1] * y1v[2],
                           x1c[2] * y1v[0], x1c[2] * y1v[1], x1c[2] * y1v[2] };
            float p[5];
            proj5(O, p);
#pragma unroll
            for (int m = 0; m < 5; ++m) t112v[m][cc] = p[m];
            float Mx2[9], P[9];
            mat5(x2c, Mx2);
            mm3(Mx2, My2, P);      // proj(Mx2*My2) == sym. A222 contraction
            proj5(P, p);
#pragma unroll
            for (int m = 0; m < 5; ++m) t222v[m][cc] = p[m];
        }
#pragma unroll
        for (int m = 0; m < 5; ++m) {
#pragma unroll
            for (int cc = 0; cc < 8; ++cc) {
                const int c = cb + cc;
                aS[el * 96 + 0  + c] = (_Float16)(x0r[cc] * y2v[m] * wt[2 * 16 + c]);
                aS[el * 96 + 16 + c] = (_Float16)(t112v[m][cc] * wt[5 * 16 + c]);
                aS[el * 96 + 32 + c] = (_Float16)(x2r[m][cc] * wt[7 * 16 + c]);
                aS[el * 96 + 48 + c] = (_Float16)(t222v[m][cc] * wt[10 * 16 + c]);
            }
            WSYNC();
            f8 na = zero8();
#pragma unroll
            for (int kc = 0; kc < 2; ++kc)
                na = wmma(afrag(aS, 96, kc * 32, lane),
                          bfrag(wl + L_LIN2, 64, kc * 32, lane), na);
#pragma unroll
            for (int r = 0; r < 8; ++r)
                nS[(4 + m) * 256 + (r + 8 * g) * 16 + el] = (_Float16)(na[r] * i64);
            WSYNC();
        }

        // ---- hid = silu([scal | n0] @ Wm1 + bm1)   (K = 96: 64 from sS,
        //      32 from the comb strip at aS+64) ------------------------------
#pragma unroll
        for (int nt = 0; nt < 4; ++nt) acc[nt] = zero8();
#pragma unroll
        for (int kc = 0; kc < 2; ++kc) {
            h16 a = afrag(sS, 64, kc * 32, lane);
#pragma unroll
            for (int nt = 0; nt < 4; ++nt)
                acc[nt] = wmma(a, bfrag(wl + L_M1 + nt * 16 * 96, 96, kc * 32, lane), acc[nt]);
        }
        {
            h16 a = afrag(aS + 64, 96, 0, lane);
#pragma unroll
            for (int nt = 0; nt < 4; ++nt)
                acc[nt] = wmma(a, bfrag(wl + L_M1 + nt * 16 * 96, 96, 64, lane), acc[nt]);
        }
#pragma unroll
        for (int nt = 0; nt < 4; ++nt) {
            const float bb = bm1[l * 64 + nt * 16 + el];
#pragma unroll
            for (int r = 0; r < 8; ++r)
                aS[(r + 8 * g) * 96 + nt * 16 + el] = (_Float16)silu_f(acc[nt][r] + bb);
        }
        WSYNC();

        // ---- scal = hid @ Wm2 + bm2  (also the layer output) --------------
#pragma unroll
        for (int nt = 0; nt < 4; ++nt) acc[nt] = zero8();
#pragma unroll
        for (int kc = 0; kc < 2; ++kc) {
            h16 a = afrag(aS, 96, kc * 32, lane);
#pragma unroll
            for (int nt = 0; nt < 4; ++nt)
                acc[nt] = wmma(a, bfrag(wl + L_M2 + nt * 16 * 64, 64, kc * 32, lane), acc[nt]);
        }
#pragma unroll
        for (int nt = 0; nt < 4; ++nt) {
            const float bb = bm2[l * 64 + nt * 16 + el];
#pragma unroll
            for (int r = 0; r < 8; ++r) {
                const float v = acc[nt][r] + bb;
                sS[(r + 8 * g) * 64 + nt * 16 + el] = (_Float16)v;
                out[((size_t)(e0 + r + 8 * g) * 2 + l) * 64 + nt * 16 + el] = v;
            }
        }
        WSYNC();

        // ---- env = scal @ Wenv + benv -------------------------------------
        {
            f8 ea = zero8();
#pragma unroll
            for (int kc = 0; kc < 2; ++kc)
                ea = wmma(afrag(sS, 64, kc * 32, lane),
                          bfrag(wl + L_ENV, 64, kc * 32, lane), ea);
            const float bb = benv[l * 16 + el];
#pragma unroll
            for (int r = 0; r < 8; ++r) envS[(r + 8 * g) * 16 + el] = ea[r] + bb;
        }
        WSYNC();

        // ---- x = n * env (gated features for next layer) ------------------
#pragma unroll
        for (int cc = 0; cc < 8; ++cc) {
            envr[cc] = envS[el * 16 + cb + cc];
            x0r[cc] = (float)nS[0 * 256 + el * 16 + cb + cc] * envr[cc];
#pragma unroll
            for (int i = 0; i < 3; ++i)
                x1r[i][cc] = (float)nS[(1 + i) * 256 + el * 16 + cb + cc] * envr[cc];
#pragma unroll
            for (int m = 0; m < 5; ++m)
                x2r[m][cc] = (float)nS[(4 + m) * 256 + el * 16 + cb + cc] * envr[cc];
        }
    }
}

extern "C" void kernel_launch(void* const* d_in, const int* in_sizes, int n_in,
                              void* d_out, int out_size, void* d_ws, size_t ws_size,
                              hipStream_t stream) {
    const float* bond_dist = (const float*)d_in[0];
    const float* bond_diff = (const float*)d_in[1];
    const float* emb       = (const float*)d_in[2];
    const float* W_e1      = (const float*)d_in[3];
    const float* b_e1      = (const float*)d_in[4];
    const float* W_e2      = (const float*)d_in[5];
    const float* b_e2      = (const float*)d_in[6];
    const float* W_env_e   = (const float*)d_in[7];
    const float* b_env_e   = (const float*)d_in[8];
    const float* w_tp      = (const float*)d_in[9];
    const float* Wlin0     = (const float*)d_in[10];
    const float* Wlin1     = (const float*)d_in[11];
    const float* Wlin2     = (const float*)d_in[12];
    const float* Wm1       = (const float*)d_in[13];
    const float* bm1       = (const float*)d_in[14];
    const float* Wm2       = (const float*)d_in[15];
    const float* bm2       = (const float*)d_in[16];
    const float* Wenv      = (const float*)d_in[17];
    const float* benv      = (const float*)d_in[18];
    const int*   Z         = (const int*)d_in[19];
    const int*   ei        = (const int*)d_in[20];
    float* out = (float*)d_out;

    const int E = in_sizes[0];
    const int blocks = E / 64;   // 4 waves x 16 edges per block
    allegro_fwd<<<blocks, 128, SMEM_B, stream>>>(
        bond_dist, bond_diff, emb, W_e1, b_e1, W_e2, b_e2, W_env_e, b_env_e,
        w_tp, Wlin0, Wlin1, Wlin2, Wm1, bm1, Wm2, bm2, Wenv, benv, Z, ei, out);
}
